// ResonatorRouter_84181359001987
// MI455X (gfx1250) — matrix operand order repaired and checked by reading, not compiled, not run on hardware
//
#include <hip/hip_runtime.h>
#include <math.h>

#define NUM_EXPERTS 64
#define HIDDEN 4096
#define DIM 4096
#define NROWS 8192
#define RES_ITERS 3
#define SHARP 10.0f
#define NBINS 2049            // DIM/2 + 1
#define EPITCH 4104           // padded floats per expert-major spectrum row
#define TPITCH 128            // floats per bin in bin-major transposed Ef (64 complex)

typedef __attribute__((ext_vector_type(16))) __bf16 v16bf;
typedef __attribute__((ext_vector_type(8)))  float  v8f;

union Frag { v16bf v; unsigned u[8]; };

// float->bf16 casts are round-to-nearest-even; let the backend use the packed
// convert hardware instead of manual bit-twiddling.
__device__ __forceinline__ unsigned pack_bf16(float a, float b) {
  union { __bf16 h[2]; unsigned u; } r;
  r.h[0] = (__bf16)a;
  r.h[1] = (__bf16)b;
  return r.u;
}

// ---------------------------------------------------------------------------
// In-LDS 4096-point complex radix-2 DIT FFT, 256 threads.
// ---------------------------------------------------------------------------
__device__ void fft4096(float2* s) {
  const int tid = threadIdx.x;
  for (int i = tid; i < DIM; i += 256) {
    unsigned r = __brev((unsigned)i) >> 20;          // 12-bit reversal
    if ((int)r > i) { float2 t = s[i]; s[i] = s[r]; s[r] = t; }
  }
  __syncthreads();
  for (int len = 2; len <= DIM; len <<= 1) {
    const int half = len >> 1;
    const float base = -6.283185307179586f / (float)len;
    for (int b = tid; b < DIM / 2; b += 256) {
      int grp = b / half;
      int pos = b - grp * half;
      int i0 = grp * len + pos;
      int i1 = i0 + half;
      float sn, c;
      __sincosf(base * (float)pos, &sn, &c);
      float2 u = s[i0], v = s[i1];
      float vr = v.x * c - v.y * sn;
      float vi = v.x * sn + v.y * c;
      s[i0] = make_float2(u.x + vr, u.y + vi);
      s[i1] = make_float2(u.x - vr, u.y - vi);
    }
    __syncthreads();
  }
}

// Load a length-4096 f32 row, L2-normalize (max(norm,1e-12)), into LDS {x,0}.
__device__ void load_normalized(const float* __restrict__ src, float2* buf, float* red) {
  const int tid = threadIdx.x;
  float ss = 0.f;
  for (int i = tid; i < DIM; i += 256) {
    float v = src[i];
    buf[i] = make_float2(v, 0.f);
    ss += v * v;
  }
  red[tid] = ss;
  __syncthreads();
  for (int s = 128; s > 0; s >>= 1) {
    if (tid < s) red[tid] += red[tid + s];
    __syncthreads();
  }
  float scale = 1.f / fmaxf(sqrtf(red[0]), 1e-12f);
  for (int i = tid; i < DIM; i += 256) buf[i].x *= scale;
  __syncthreads();
}

// ---------------------------------------------------------------------------
// Kernel 1: per-expert spectra. Ef (expert-major), Ewf (Parseval-weighted),
// EfT (bin-major transpose for the resonator's k-contiguous reads), Lf.
// ---------------------------------------------------------------------------
__global__ __launch_bounds__(256)
void expert_setup(const float* __restrict__ labels, const float* __restrict__ sigs,
                  float* __restrict__ Ef, float* __restrict__ Ewf,
                  float* __restrict__ EfT, float* __restrict__ Lf) {
  extern __shared__ float smem[];
  float2* buf = (float2*)smem;              // 4096 complex
  float*  red = smem + 2 * DIM;             // 256
  const int k = blockIdx.x;
  const int tid = threadIdx.x;

  load_normalized(sigs + (size_t)k * DIM, buf, red);
  fft4096(buf);
  for (int b = tid; b < NBINS; b += 256) {
    float2 X = buf[b];
    float w = (b == 0 || b == DIM / 2) ? 1.f : 2.f;
    Ef [(size_t)k * EPITCH + 2 * b]     = X.x;
    Ef [(size_t)k * EPITCH + 2 * b + 1] = X.y;
    Ewf[(size_t)k * EPITCH + 2 * b]     = w * X.x;
    Ewf[(size_t)k * EPITCH + 2 * b + 1] = w * X.y;
    EfT[(size_t)b * TPITCH + 2 * k]     = X.x;
    EfT[(size_t)b * TPITCH + 2 * k + 1] = X.y;
  }
  __syncthreads();

  load_normalized(labels + (size_t)k * DIM, buf, red);
  fft4096(buf);
  for (int b = tid; b < NBINS; b += 256) {
    float2 X = buf[b];
    Lf[(size_t)k * EPITCH + 2 * b]     = X.x;
    Lf[(size_t)k * EPITCH + 2 * b + 1] = X.y;
  }
}

// ---------------------------------------------------------------------------
// Kernel 2: R_f[b] = sum_k Ef_k[b] * Lf_k[b]   (bind == spectral product)
// ---------------------------------------------------------------------------
__global__ __launch_bounds__(256)
void build_rf(const float* __restrict__ Ef, const float* __restrict__ Lf,
              float* __restrict__ Rf) {
  int b = blockIdx.x * blockDim.x + threadIdx.x;
  if (b >= NBINS) return;
  float rr = 0.f, ri = 0.f;
  for (int k = 0; k < NUM_EXPERTS; ++k) {
    const size_t o = (size_t)k * EPITCH + 2 * b;
    float er = Ef[o], ei = Ef[o + 1];
    float lr = Lf[o], li = Lf[o + 1];
    rr += er * lr - ei * li;
    ri += er * li + ei * lr;
  }
  Rf[2 * b] = rr;
  Rf[2 * b + 1] = ri;
}

// ---------------------------------------------------------------------------
// Kernel 3: Y = x @ W^T  (M=8192, N=4096, K=4096) via bf16 WMMA.
// Block tile 128x128, K-step 32, register-staged DOUBLE-BUFFERED LDS:
// while the 8 WMMAs of step i consume LDS[cur], the f32 tile for step i+1 is
// already in flight to VGPRs; it is packed to bf16 into LDS[cur^1] between
// the two barriers. A(16x32)/B(32x16) fragments use contiguous k = 16*hi + s,
// the same order in both operands (not the ISA table order).
// ---------------------------------------------------------------------------
#define BM 128
#define BN 128
#define BK 32
#define LDP 17    // 16 packed-bf16 dwords per row + 1 pad (bank-conflict break)

__global__ __launch_bounds__(256)
void gemm_xwT_bf16(const float* __restrict__ X, const float* __restrict__ W,
                   float* __restrict__ Y) {
  __shared__ unsigned As[2][BM * LDP];
  __shared__ unsigned Bs[2][BN * LDP];
  const int tid  = threadIdx.x;
  const int lane = tid & 31;
  const int wave = tid >> 5;
  const int m0 = blockIdx.y * BM;
  const int n0 = blockIdx.x * BN;
  const int mwr = wave >> 1;          // 0..3 : which 32-row strip
  const int wc  = wave & 1;           // 0..1 : which 64-col strip
  const int hi  = lane >> 4;          // half-wave select
  const int l15 = lane & 15;

  // This thread's 4 staging chunks (row / float4-within-row are loop-invariant)
  int srow[4], sc4[4];
#pragma unroll
  for (int i = 0; i < 4; ++i) {
    int f = tid + 256 * i;            // 0..1023 float4 chunks per tile
    srow[i] = f >> 3;
    sc4[i]  = f & 7;
  }

  float4 ra[4], rb[4];
  auto load_tile = [&](int k0) {
#pragma unroll
    for (int i = 0; i < 4; ++i) {
      ra[i] = *(const float4*)(X + (size_t)(m0 + srow[i]) * HIDDEN + k0 + sc4[i] * 4);
      rb[i] = *(const float4*)(W + (size_t)(n0 + srow[i]) * HIDDEN + k0 + sc4[i] * 4);
    }
  };
  auto store_tile = [&](int bi) {
#pragma unroll
    for (int i = 0; i < 4; ++i) {
      unsigned* ap = &As[bi][srow[i] * LDP + sc4[i] * 2];
      ap[0] = pack_bf16(ra[i].x, ra[i].y);
      ap[1] = pack_bf16(ra[i].z, ra[i].w);
      unsigned* bp = &Bs[bi][srow[i] * LDP + sc4[i] * 2];
      bp[0] = pack_bf16(rb[i].x, rb[i].y);
      bp[1] = pack_bf16(rb[i].z, rb[i].w);
    }
  };

  const v8f vzero = {0.f, 0.f, 0.f, 0.f, 0.f, 0.f, 0.f, 0.f};
  v8f acc[2][4];
#pragma unroll
  for (int i = 0; i < 2; ++i)
#pragma unroll
    for (int j = 0; j < 4; ++j) acc[i][j] = vzero;

  load_tile(0);
  store_tile(0);
  __syncthreads();

  int cur = 0;
  for (int k0 = 0; k0 < HIDDEN; k0 += BK) {
    const bool more = (k0 + BK) < HIDDEN;
    if (more) load_tile(k0 + BK);     // global loads overlap the WMMAs below

    Frag a[2], b[4];
#pragma unroll
    for (int i = 0; i < 2; ++i) {
      const unsigned* row = &As[cur][(mwr * 32 + i * 16 + l15) * LDP];
#pragma unroll
      for (int j = 0; j < 8; ++j) {
        // A 16x32 bf16: vgpr j -> K=2j+16*hi (matches B)
        int u = j + 8 * hi;
        a[i].u[j] = row[u];
      }
    }
#pragma unroll
    for (int j = 0; j < 4; ++j) {
      const unsigned* row = &Bs[cur][(wc * 64 + j * 16 + l15) * LDP];
#pragma unroll
      for (int q = 0; q < 8; ++q)
        b[j].u[q] = row[q + 8 * hi];  // B 32x16: lanes<16 K=0..15, lanes>=16 K=16..31
    }
#pragma unroll
    for (int i = 0; i < 2; ++i)
#pragma unroll
      for (int j = 0; j < 4; ++j)
        acc[i][j] = __builtin_amdgcn_wmma_f32_16x16x32_bf16(
            false, a[i].v, false, b[j].v, (short)0, acc[i][j], false, false);

    if (more) {
      __syncthreads();                // everyone done reading LDS[cur^1] (previous step)
      store_tile(cur ^ 1);
      __syncthreads();                // writes visible before next compute
      cur ^= 1;
    }
  }

  // C/D 16x16 f32 layout: vgpr r -> M = r + 8*hi, N = lane&15.
#pragma unroll
  for (int i = 0; i < 2; ++i)
#pragma unroll
    for (int j = 0; j < 4; ++j)
#pragma unroll
      for (int r = 0; r < 8; ++r) {
        int mrow = m0 + mwr * 32 + i * 16 + r + 8 * hi;
        int ncol = n0 + wc * 64 + j * 16 + l15;
        Y[(size_t)mrow * DIM + ncol] = acc[i][j][r];
      }
}

// ---------------------------------------------------------------------------
// Kernel 4: per-row resonator, entirely in the frequency domain.
//   B = rfft(x_proj) * conj(R_f);  Z = B;  repeat 3x:
//     sim_k = Parseval<Z,Ewf_k> / ||z||;  attn = softmax(10*sim)
//     Z = B * conj( sum_k attn_k * Ef_k )
//   scores = final sims; top-2 + softmax -> weights/indices.
// ---------------------------------------------------------------------------
__global__ __launch_bounds__(256)
void resonate(const float* __restrict__ Y, const float* __restrict__ EfT,
              const float* __restrict__ Ewf, const float* __restrict__ Rf,
              float* __restrict__ out_w, int* __restrict__ out_i,
              float* __restrict__ out_s) {
  extern __shared__ float smem[];
  float2* buf  = (float2*)smem;                       // 4096 complex (FFT)
  float2* Zf   = buf;                                 // alias: first 2049 bins
  float2* Bf   = (float2*)(smem + 2 * DIM);           // 2049 complex
  float*  red  = smem + 2 * DIM + 4100;               // 256
  float*  sims = red + 256;                           // 64
  float*  attn = sims + 64;                           // 64

  const int row  = blockIdx.x;
  const int tid  = threadIdx.x;
  const int lane = tid & 31;
  const int wave = tid >> 5;

  load_normalized(Y + (size_t)row * DIM, buf, red);   // x_proj row
  fft4096(buf);
  for (int b = tid; b < NBINS; b += 256) {
    float2 X = buf[b];
    float rr = Rf[2 * b], ri = Rf[2 * b + 1];
    float2 B = make_float2(X.x * rr + X.y * ri,       // X * conj(R)
                           X.y * rr - X.x * ri);
    Bf[b] = B;
    Zf[b] = B;                                        // initial z = unbind(x, R)
  }
  __syncthreads();

  float invn = 0.f;
  for (int it = 0; it <= RES_ITERS; ++it) {
    // Per-wave expert dot products (8 experts per wave, lanes stride bins).
#pragma unroll
    for (int kk = 0; kk < 8; ++kk) {
      int k = wave * 8 + kk;
      const float2* e = (const float2*)(Ewf + (size_t)k * EPITCH);
      float acc = 0.f;
      for (int b = lane; b < NBINS; b += 32) {
        float2 Z = Zf[b];
        float2 ev = e[b];
        acc += Z.x * ev.x + Z.y * ev.y;
      }
#pragma unroll
      for (int off = 16; off > 0; off >>= 1)
        acc += __shfl_down(acc, off, 32);              // wave32 reduction
      if (lane == 0) sims[k] = acc;
    }
    // ||z||^2 via Parseval (unscaled by 1/N; folded below).
    float z2 = 0.f;
    for (int b = tid; b < NBINS; b += 256) {
      float2 Z = Zf[b];
      float w = (b == 0 || b == DIM / 2) ? 1.f : 2.f;
      z2 += w * (Z.x * Z.x + Z.y * Z.y);
    }
    red[tid] = z2;
    __syncthreads();
    for (int s = 128; s > 0; s >>= 1) {
      if (tid < s) red[tid] += red[tid + s];
      __syncthreads();
    }
    invn = rsqrtf(fmaxf(red[0] * (float)DIM, 1e-24f)); // sim = dot * invn
    __syncthreads();

    if (it == RES_ITERS) break;

    if (tid < 64) {                                    // softmax(10 * sim)
      float mx = -1e30f;
      for (int j = 0; j < 64; ++j) mx = fmaxf(mx, sims[j]);
      float se = 0.f;
      for (int j = 0; j < 64; ++j) se += __expf(SHARP * invn * (sims[j] - mx));
      attn[tid] = __expf(SHARP * invn * (sims[tid] - mx)) / se;
    }
    __syncthreads();

    // Zc_f = attn @ Ef (bin-major EfT: 64 contiguous complex per bin);
    // Z = B * conj(Zc)
    for (int b = tid; b < NBINS; b += 256) {
      const float2* e = (const float2*)(EfT + (size_t)b * TPITCH);
      float zcr = 0.f, zci = 0.f;
#pragma unroll 8
      for (int k = 0; k < NUM_EXPERTS; ++k) {
        float a = attn[k];
        float2 ev = e[k];
        zcr += a * ev.x;
        zci += a * ev.y;
      }
      float2 B = Bf[b];
      Zf[b] = make_float2(B.x * zcr + B.y * zci,
                          B.y * zcr - B.x * zci);
    }
    __syncthreads();
  }

  if (tid < 64) {
    float s = sims[tid] * invn;                        // final cosine scores
    out_s[(size_t)row * 64 + tid] = s;
    sims[tid] = s;
  }
  __syncthreads();
  if (tid == 0) {
    int i1 = 0; float s1 = sims[0];
    for (int j = 1; j < 64; ++j) if (sims[j] > s1) { s1 = sims[j]; i1 = j; }
    int i2 = -1; float s2 = -1e30f;
    for (int j = 0; j < 64; ++j) if (j != i1 && sims[j] > s2) { s2 = sims[j]; i2 = j; }
    float w1 = 1.f / (1.f + __expf(s2 - s1));          // softmax over top-2, T=1
    out_w[(size_t)row * 2 + 0] = w1;
    out_w[(size_t)row * 2 + 1] = 1.f - w1;
    out_i[(size_t)row * 2 + 0] = i1;
    out_i[(size_t)row * 2 + 1] = i2;
  }
}

// ---------------------------------------------------------------------------
extern "C" void kernel_launch(void* const* d_in, const int* in_sizes, int n_in,
                              void* d_out, int out_size, void* d_ws, size_t ws_size,
                              hipStream_t stream) {
  const float* x      = (const float*)d_in[0];   // [8192, 4096]
  const float* W      = (const float*)d_in[1];   // [4096, 4096]
  const float* labels = (const float*)d_in[2];   // [64, 4096]
  const float* sigs   = (const float*)d_in[3];   // [64, 4096]

  float* ws  = (float*)d_ws;
  float* Ef  = ws;                                       // 64 * EPITCH
  float* Ewf = Ef  + (size_t)NUM_EXPERTS * EPITCH;       // 64 * EPITCH
  float* Lf  = Ewf + (size_t)NUM_EXPERTS * EPITCH;       // 64 * EPITCH
  float* EfT = Lf  + (size_t)NUM_EXPERTS * EPITCH;       // NBINS * TPITCH
  float* Rf  = EfT + (size_t)NBINS * TPITCH;             // EPITCH
  float* Yp  = Rf  + EPITCH;                             // 8192 * 4096

  float* out_w = (float*)d_out;                          // [8192, 2]
  int*   out_i = (int*)((float*)d_out + NROWS * 2);      // [8192, 2] (int bits)
  float* out_s = (float*)d_out + NROWS * 4;              // [8192, 64]

  const size_t smem_setup = (size_t)(2 * DIM + 256) * sizeof(float);                  // 33.8 KB
  const size_t smem_res   = (size_t)(2 * DIM + 4100 + 256 + 64 + 64) * sizeof(float); // 50.7 KB

  expert_setup<<<dim3(NUM_EXPERTS), dim3(256), smem_setup, stream>>>(labels, sigs, Ef, Ewf, EfT, Lf);
  build_rf<<<dim3((NBINS + 255) / 256), dim3(256), 0, stream>>>(Ef, Lf, Rf);
  gemm_xwT_bf16<<<dim3(DIM / BN, NROWS / BM), dim3(256), 0, stream>>>(x, W, Yp);
  resonate<<<dim3(NROWS), dim3(256), smem_res, stream>>>(Yp, EfT, Ewf, Rf, out_w, out_i, out_s);
}
